// LightGCNBase_1589137900099
// MI455X (gfx1250) — matrix operand order repaired and checked by reading, not compiled, not run on hardware
//
#include <hip/hip_runtime.h>
#include <hip/hip_fp16.h>

typedef __attribute__((ext_vector_type(16))) _Float16 v16h;
typedef __attribute__((ext_vector_type(8)))  _Float16 h8;
typedef __attribute__((ext_vector_type(8)))  float    v8f;

#define NU 100000
#define NI 50000
#define NN 150000
#define NE 2400000
#define DD 64
#define NB 4096
#define KP1 320   /* padded K for eeg (310) and combined (304) GEMMs */

// ---------------------------------------------------------------------------
// ego = concat(user_emb, item_emb); acc = ego
// ---------------------------------------------------------------------------
__global__ void init_ego_kernel(const float* __restrict__ ue, const float* __restrict__ ie,
                                float* __restrict__ ego, float* __restrict__ acc) {
  long long i = (long long)blockIdx.x * blockDim.x + threadIdx.x;
  const long long tot = (long long)NN * DD;
  if (i >= tot) return;
  const long long usz = (long long)NU * DD;
  float v = (i < usz) ? ue[i] : ie[i - usz];
  ego[i] = v;
  acc[i] = v;
}

__global__ void zero_kernel(float* __restrict__ p, long long n) {
  long long i = (long long)blockIdx.x * blockDim.x + threadIdx.x;
  if (i < n) p[i] = 0.f;
}

__global__ void acc_add_kernel(float* __restrict__ acc, const float* __restrict__ nxt,
                               long long n) {
  long long i = (long long)blockIdx.x * blockDim.x + threadIdx.x;
  if (i < n) acc[i] += nxt[i];
}

// ---------------------------------------------------------------------------
// Edge-parallel SpMM scatter: nxt[row] += val * ego[col]
// 16 lanes/edge, float4 gather (global_load_b128, L2-resident: 38.4MB table
// vs 192MB L2), native f32 atomic scatter.
// ---------------------------------------------------------------------------
__global__ void spmm_kernel(const int* __restrict__ rows, const int* __restrict__ cols,
                            const float* __restrict__ vals, const float* __restrict__ ego,
                            float* __restrict__ nxt) {
  long long t = (long long)blockIdx.x * blockDim.x + threadIdx.x;
  int e = (int)(t >> 4);
  if (e >= NE) return;
  int q = ((int)t & 15) << 2;
  int r = rows[e];
  int c = cols[e];
  float v = vals[e];
  const float4 s = *reinterpret_cast<const float4*>(ego + (long long)c * DD + q);
  float* dst = nxt + (long long)r * DD + q;
  unsafeAtomicAdd(dst + 0, v * s.x);
  unsafeAtomicAdd(dst + 1, v * s.y);
  unsafeAtomicAdd(dst + 2, v * s.z);
  unsafeAtomicAdd(dst + 3, v * s.w);
}

// ---------------------------------------------------------------------------
// Prep: pad+convert eeg (B x 310 f32) -> (B x 320 f16), zero-padded K.
// ---------------------------------------------------------------------------
__global__ void pad_eeg_kernel(const float* __restrict__ eeg, _Float16* __restrict__ out) {
  int i = blockIdx.x * blockDim.x + threadIdx.x;
  if (i >= NB * KP1) return;
  int row = i / KP1, col = i - row * KP1;
  out[i] = (col < 310) ? (_Float16)eeg[row * 310 + col] : (_Float16)0.f;
}

// Prep: W (K x N f32, row-major) -> Wt (N x Kp f16, K-major, zero padded).
__global__ void wt_kernel(const float* __restrict__ W, _Float16* __restrict__ Wt,
                          int K, int N, int Kp) {
  int i = blockIdx.x * blockDim.x + threadIdx.x;
  if (i >= N * Kp) return;
  int n = i / Kp, k = i - n * Kp;
  Wt[i] = (k < K) ? (_Float16)W[k * N + n] : (_Float16)0.f;
}

// ---------------------------------------------------------------------------
// Gather u/i embeddings into outputs (f32) and combined (f16, row length 320):
// cols [0:64)=u, [64:128)=i, [128:144)=eeg16 (written by gemm3),
// [144:304)=ctx16, [304:320)=zero pad.
// ---------------------------------------------------------------------------
__global__ void gather_kernel(const float* __restrict__ acc, const int* __restrict__ user_id,
                              const int* __restrict__ item_id, const float* __restrict__ ctx,
                              const float* __restrict__ cW, const float* __restrict__ cb,
                              _Float16* __restrict__ comb, float* __restrict__ u_v,
                              float* __restrict__ i_v) {
  int b = blockIdx.x;
  int t = threadIdx.x;                       // block of 320 threads
  _Float16* crow = comb + (long long)b * KP1;
  if (t < 64) {
    float u = acc[(long long)user_id[b] * DD + t] * 0.25f;   // /(L+1)
    crow[t] = (_Float16)u;
    u_v[(long long)b * 64 + t] = u;
  } else if (t < 128) {
    int d = t - 64;
    float x = acc[((long long)NU + item_id[b]) * DD + d] * 0.25f;
    crow[64 + d] = (_Float16)x;
    i_v[(long long)b * 64 + d] = x;
  } else if (t < 288) {
    int d = t - 128;                         // 0..159
    int j = d >> 4, cc = d & 15;
    crow[144 + d] = (_Float16)(ctx[b * 10 + j] * cW[cc] + cb[cc]);
  } else if (t < 304) {
    crow[t + 16] = (_Float16)0.f;            // cols 304..319 zero pad
  }
}

// ---------------------------------------------------------------------------
// WMMA GEMM, one wave per 16x16 tile. A: f16 row-major (M x KP). B: f16
// K-major (N x KP). KP compile-time multiple of 32 -> no bounds checks;
// each fragment is contiguous 16B chunks -> global_load_b128 + v_wmma only.
// Fragment layouts per CDNA5 ISA 7.12.2 (wave32):
//   A lane L: row M=L%16, elems j -> K = half*8 + j (+16 for j>=8)
//   B lane L: col N=L%16, elems j -> K = half*8 + j (+16 for j>=8)
//   D lane L: col N=L%16, vgpr v -> row M = 8*half + v
// ---------------------------------------------------------------------------
template <int KP>
__global__ void wmma_gemm_kernel(const _Float16* __restrict__ X, const _Float16* __restrict__ Wt,
                                 const float* __restrict__ bias, _Float16* __restrict__ Out,
                                 int M, int N, int ldo, int relu) {
  int wave = (int)(((long long)blockIdx.x * blockDim.x + threadIdx.x) >> 5);
  int lane = threadIdx.x & 31;
  int tilesN = N >> 4;
  int tm = wave / tilesN;
  int tn = wave - tm * tilesN;
  if (tm * 16 >= M) return;                  // uniform per wave
  const int half = lane >> 4;
  const int mn = lane & 15;
  const int ncol = tn * 16 + mn;

  const _Float16* arow = X + (long long)(tm * 16 + mn) * KP + half * 8;
  const _Float16* bcol = Wt + (long long)ncol * KP + half * 8;

  v8f c = {0.f, 0.f, 0.f, 0.f, 0.f, 0.f, 0.f, 0.f};
#pragma unroll
  for (int k0 = 0; k0 < KP; k0 += 32) {
    h8 a0 = *reinterpret_cast<const h8*>(arow + k0);        // K = k0+half*8 + 0..7
    h8 a1 = *reinterpret_cast<const h8*>(arow + k0 + 16);   // K = k0+16+half*8 + 0..7
    h8 b0 = *reinterpret_cast<const h8*>(bcol + k0);        // K = k0+half*8 + 0..7
    h8 b1 = *reinterpret_cast<const h8*>(bcol + k0 + 16);   // K = k0+16+half*8 + 0..7
    v16h a = __builtin_shufflevector(a0, a1, 0, 1, 2, 3, 4, 5, 6, 7,
                                     8, 9, 10, 11, 12, 13, 14, 15);
    v16h b = __builtin_shufflevector(b0, b1, 0, 1, 2, 3, 4, 5, 6, 7,
                                     8, 9, 10, 11, 12, 13, 14, 15);
    c = __builtin_amdgcn_wmma_f32_16x16x32_f16(false, a, false, b, (short)0, c,
                                               false, false);
  }

  float bv = bias[ncol];
#pragma unroll
  for (int v = 0; v < 8; ++v) {
    float r = c[v] + bv;
    if (relu) r = fmaxf(r, 0.f);
    int m = tm * 16 + half * 8 + v;
    Out[(long long)m * ldo + ncol] = (_Float16)r;
  }
}

// Final N=1 projection: pred[b] = F1[b,:] . fW2 + fb2
__global__ void final_dot_kernel(const _Float16* __restrict__ F1, const float* __restrict__ fW2,
                                 const float* __restrict__ fb2, float* __restrict__ pred) {
  int b = blockIdx.x * blockDim.x + threadIdx.x;
  if (b >= NB) return;
  float s = fb2[0];
#pragma unroll 8
  for (int k = 0; k < 64; ++k) s += (float)F1[(long long)b * 64 + k] * fW2[k];
  pred[b] = s;
}

extern "C" void kernel_launch(void* const* d_in, const int* in_sizes, int n_in,
                              void* d_out, int out_size, void* d_ws, size_t ws_size,
                              hipStream_t stream) {
  (void)in_sizes; (void)n_in; (void)out_size; (void)ws_size;

  const float* user_emb = (const float*)d_in[0];
  const float* item_emb = (const float*)d_in[1];
  const int*   adj_rows = (const int*)d_in[2];
  const int*   adj_cols = (const int*)d_in[3];
  const float* adj_vals = (const float*)d_in[4];
  const int*   user_id  = (const int*)d_in[5];
  const int*   item_id  = (const int*)d_in[6];
  const float* eeg      = (const float*)d_in[7];
  const float* ctx      = (const float*)d_in[8];
  const float* eW1 = (const float*)d_in[9];
  const float* eb1 = (const float*)d_in[10];
  const float* eW2 = (const float*)d_in[11];
  const float* eb2 = (const float*)d_in[12];
  const float* eW3 = (const float*)d_in[13];
  const float* eb3 = (const float*)d_in[14];
  const float* cW  = (const float*)d_in[15];
  const float* cb  = (const float*)d_in[16];
  const float* fW1 = (const float*)d_in[17];
  const float* fb1 = (const float*)d_in[18];
  const float* fW2 = (const float*)d_in[19];
  const float* fb2 = (const float*)d_in[20];

  // Outputs: prediction (4096) | u_v (4096*64) | i_v (4096*64)
  float* pred_out = (float*)d_out;
  float* u_v_out  = pred_out + NB;
  float* i_v_out  = u_v_out + (long long)NB * 64;

  // ---- workspace carve-up (all chunks 16B aligned) ----
  char* p = (char*)d_ws;
  auto take = [&](size_t bytes) -> char* {
    char* r = p;
    p += (bytes + 15) & ~(size_t)15;
    return r;
  };
  const long long EMB = (long long)NN * DD;                  // 9.6M floats
  float*    bufA   = (float*)take(EMB * 4);
  float*    bufB   = (float*)take(EMB * 4);
  float*    acc    = (float*)take(EMB * 4);
  _Float16* eeg_h  = (_Float16*)take((size_t)NB * KP1 * 2);  // 4096x320 f16
  _Float16* comb_h = (_Float16*)take((size_t)NB * KP1 * 2);  // 4096x320 f16
  _Float16* H1_h   = (_Float16*)take((size_t)NB * 64 * 2);
  _Float16* H2_h   = (_Float16*)take((size_t)NB * 32 * 2);
  _Float16* F1_h   = (_Float16*)take((size_t)NB * 64 * 2);
  _Float16* eW1t   = (_Float16*)take((size_t)64 * KP1 * 2);  // 64 x 320
  _Float16* eW2t   = (_Float16*)take((size_t)32 * 64 * 2);   // 32 x 64
  _Float16* eW3t   = (_Float16*)take((size_t)16 * 32 * 2);   // 16 x 32
  _Float16* fW1t   = (_Float16*)take((size_t)64 * KP1 * 2);  // 64 x 320

  const int ethreads = 256;
  const int eblocks  = (int)((EMB + ethreads - 1) / ethreads);

  // 1) ego = concat(user, item); acc = ego
  init_ego_kernel<<<eblocks, ethreads, 0, stream>>>(user_emb, item_emb, bufA, acc);

  // Prep (independent of graph phase): pad/convert activations + weights
  pad_eeg_kernel<<<(NB * KP1 + 255) / 256, 256, 0, stream>>>(eeg, eeg_h);
  wt_kernel<<<(64 * KP1 + 255) / 256, 256, 0, stream>>>(eW1, eW1t, 310, 64, KP1);
  wt_kernel<<<(32 * 64 + 255) / 256, 256, 0, stream>>>(eW2, eW2t, 64, 32, 64);
  wt_kernel<<<(16 * 32 + 255) / 256, 256, 0, stream>>>(eW3, eW3t, 32, 16, 32);
  wt_kernel<<<(64 * KP1 + 255) / 256, 256, 0, stream>>>(fW1, fW1t, 304, 64, KP1);

  // 2) L=3 propagation layers
  float* ego = bufA;
  float* nxt = bufB;
  for (int l = 0; l < 3; ++l) {
    zero_kernel<<<eblocks, ethreads, 0, stream>>>(nxt, EMB);
    long long thr = (long long)NE * 16;
    int sb = (int)((thr + 255) / 256);
    spmm_kernel<<<sb, 256, 0, stream>>>(adj_rows, adj_cols, adj_vals, ego, nxt);
    acc_add_kernel<<<eblocks, ethreads, 0, stream>>>(acc, nxt, EMB);
    float* t = ego; ego = nxt; nxt = t;
  }

  // 3) EEG MLP via WMMA
  auto blocks_for = [](int M, int N) { return (((M / 16) * (N / 16)) + 7) / 8; };
  wmma_gemm_kernel<KP1><<<blocks_for(NB, 64), 256, 0, stream>>>(
      eeg_h, eW1t, eb1, H1_h, NB, 64, 64, 1);
  wmma_gemm_kernel<64><<<blocks_for(NB, 32), 256, 0, stream>>>(
      H1_h, eW2t, eb2, H2_h, NB, 32, 32, 1);
  wmma_gemm_kernel<32><<<blocks_for(NB, 16), 256, 0, stream>>>(
      H2_h, eW3t, eb3, comb_h + 128, NB, 16, KP1, 0);   // eeg16 -> combined cols 128..143

  // 4) Gather u/i (+outputs) and ctx16/pad into combined (f16)
  gather_kernel<<<NB, 320, 0, stream>>>(acc, user_id, item_id, ctx, cW, cb,
                                        comb_h, u_v_out, i_v_out);

  // 5) Fusion head
  wmma_gemm_kernel<KP1><<<blocks_for(NB, 64), 256, 0, stream>>>(
      comb_h, fW1t, fb1, F1_h, NB, 64, 64, 1);
  final_dot_kernel<<<(NB + 255) / 256, 256, 0, stream>>>(F1_h, fW2, fb2, pred_out);
}
